// HeteroGraphSage_40785009443638
// MI455X (gfx1250) — matrix operand order repaired and checked by reading, not compiled, not run on hardware
//
#include <hip/hip_runtime.h>
#include <hip/hip_bf16.h>
#include <math.h>

typedef __attribute__((ext_vector_type(16))) __bf16 v16bf;
typedef __attribute__((ext_vector_type(8)))  __bf16 v8bf;
typedef __attribute__((ext_vector_type(8)))  float  v8f;

#define HDIM 128

// ---------------- WMMA fragment loaders -------------------------------------

// A fragment: 16x32 (MxK) bf16 from an f32 row-major matrix (converted on the
// fly; each element is read exactly once across the whole GEMM so a pre-pass
// would not reduce work). Per ISA layout:
// lanes 0-15:  row M=lane,    K = k0+[0..7] (v0-3) and k0+[16..23] (v4-7)
// lanes 16-31: row M=lane-16, K = k0+[8..15]       and k0+[24..31]
__device__ __forceinline__ v16bf load_a_frag_f32(const float* __restrict__ A, int m0,
                                                 int ld, int k0, int lane, int maxrow) {
  const int half = lane >> 4;
  int m = m0 + (lane & 15);
  if (m > maxrow) m = maxrow;
  const float* p = A + (size_t)m * ld + k0 + half * 8;
  v16bf a;
#pragma unroll
  for (int i = 0; i < 8; ++i) a[i] = (__bf16)p[i];
#pragma unroll
  for (int i = 0; i < 8; ++i) a[8 + i] = (__bf16)p[16 + i];
  return a;
}

// B fragment: 32x16 (KxN) where B(k,n) = W[n,k], W pre-converted to bf16,
// row-major [Nout, ld]. Two aligned 16-byte vector loads per lane.
// Lanes 0-15: N=n0+lane, K=k0+[0..7],[16..23]; lanes 16-31: N=n0+lane-16, K=k0+[8..15],[24..31].
__device__ __forceinline__ v16bf load_b_frag_bf(const __bf16* __restrict__ Wb, int n0,
                                                int ld, int k0, int lane) {
  const int n = n0 + (lane & 15);
  const int k = k0 + (lane >> 4) * 8;
  const v8bf* p = (const v8bf*)(Wb + (size_t)n * ld + k);
  const v8bf lo = p[0];
  const v8bf hi = p[2];
  return __builtin_shufflevector(lo, hi, 0, 1, 2, 3, 4, 5, 6, 7,
                                 8, 9, 10, 11, 12, 13, 14, 15);
}

// ---------------- Kernel P: f32 -> bf16 weight conversion (one-shot) --------

__global__ void cvt_f32_bf16(const float* __restrict__ src, __bf16* __restrict__ dst, int n) {
  int i = blockIdx.x * blockDim.x + threadIdx.x;
  if (i < n) dst[i] = (__bf16)src[i];
}

// ---------------- Kernel 0: zero workspace accumulator ----------------------

__global__ void zero_f4(float4* __restrict__ p, int n4) {
  int i = blockIdx.x * blockDim.x + threadIdx.x;
  if (i < n4) {
    float4 z; z.x = 0.f; z.y = 0.f; z.z = 0.f; z.w = 0.f;
    p[i] = z;
  }
}

// ---------------- Kernel 1: lin_ma = x_merchant @ Wr_ma^T + br_ma -----------
// M = NM, K = 64, N = 128. One wave -> 16x128 output strip.

__global__ void __launch_bounds__(256)
lin_ma_kernel(const float* __restrict__ Xm, const __bf16* __restrict__ Wrb,
              const float* __restrict__ br, float* __restrict__ out, int nm) {
  const int lane = threadIdx.x & 31;
  const int wave = threadIdx.x >> 5;
  const int half = lane >> 4;
  const int ntiles = (nm + 15) >> 4;
  int tile = blockIdx.x * (blockDim.x >> 5) + wave;
  if (tile >= ntiles) tile = ntiles - 1;
  const int m0 = tile << 4;

  v8f acc[8];
#pragma unroll
  for (int t = 0; t < 8; ++t) {
    const float bv = br[t * 16 + (lane & 15)];
#pragma unroll
    for (int i = 0; i < 8; ++i) acc[t][i] = bv;   // bias folded into accumulator
  }

#pragma unroll
  for (int ks = 0; ks < 64; ks += 32) {
    const v16bf a = load_a_frag_f32(Xm, m0, 64, ks, lane, nm - 1);
#pragma unroll
    for (int t = 0; t < 8; ++t) {
      const v16bf b = load_b_frag_bf(Wrb, t * 16, 64, ks, lane);
      acc[t] = __builtin_amdgcn_wmma_f32_16x16x32_bf16(
          false, a, false, b, (short)0, acc[t], false, false);
    }
  }

#pragma unroll
  for (int t = 0; t < 8; ++t) {
    const int col = t * 16 + (lane & 15);
#pragma unroll
    for (int i = 0; i < 8; ++i) {
      const int row = m0 + i + 8 * half;
      if (row < nm) out[(size_t)row * HDIM + col] = acc[t][i];
    }
  }
}

// ---------------- Kernel 2: weighted scatter-add over edges -----------------
// One wave per edge; lane handles 4 of the 128 features via f32 atomics.
// lin_ma (51 MB) and neigh_a (102 MB) are both L2-resident (192 MB L2).

__global__ void __launch_bounds__(256)
scatter_kernel(const int* __restrict__ src, const int* __restrict__ dst,
               const float* __restrict__ w, const float* __restrict__ lin,
               float* __restrict__ neigh, int ne) {
  const int lane = threadIdx.x & 31;
  const int e = blockIdx.x * (blockDim.x >> 5) + (threadIdx.x >> 5);
  if (e >= ne) return;
  const int s = src[e];
  const int d = dst[e];
  const float wv = w[e];
  const float4 v = ((const float4*)(lin + (size_t)s * HDIM))[lane];
  float* nd = neigh + (size_t)d * HDIM + lane * 4;
  __hip_atomic_fetch_add(nd + 0, wv * v.x, __ATOMIC_RELAXED, __HIP_MEMORY_SCOPE_AGENT);
  __hip_atomic_fetch_add(nd + 1, wv * v.y, __ATOMIC_RELAXED, __HIP_MEMORY_SCOPE_AGENT);
  __hip_atomic_fetch_add(nd + 2, wv * v.z, __ATOMIC_RELAXED, __HIP_MEMORY_SCOPE_AGENT);
  __hip_atomic_fetch_add(nd + 3, wv * v.w, __ATOMIC_RELAXED, __HIP_MEMORY_SCOPE_AGENT);
}

// ---------------- Kernel 3: fused proj_a -> combine -> head -----------------
// Per 16-row tile: proj GEMM (K=128), proj tile -> LDS bf16, combine GEMM
// (K=256: proj from LDS + neigh from global), ReLU, Wo reduction, sigmoid.

__global__ void __launch_bounds__(256)
account_fused(const float* __restrict__ Xa,
              const __bf16* __restrict__ Wpb, const float* __restrict__ bp,
              const __bf16* __restrict__ Wcb, const float* __restrict__ bc,
              const float* __restrict__ Wo, const float* __restrict__ bo,
              const float* __restrict__ neigh, float* __restrict__ out, int na) {
  __shared__ __bf16 lds[8 * 16 * HDIM];   // 4 KB per wave * 8 waves = 32 KB
  const int lane = threadIdx.x & 31;
  const int wave = threadIdx.x >> 5;
  const int half = lane >> 4;
  const int ntiles = (na + 15) >> 4;
  int tile = blockIdx.x * (blockDim.x >> 5) + wave;
  if (tile >= ntiles) tile = ntiles - 1;
  const int m0 = tile << 4;

  // ---- proj = Xa @ Wp^T + bp (accumulators start at bias) ----
  v8f acc[8];
#pragma unroll
  for (int t = 0; t < 8; ++t) {
    const float bv = bp[t * 16 + (lane & 15)];
#pragma unroll
    for (int i = 0; i < 8; ++i) acc[t][i] = bv;
  }
#pragma unroll
  for (int ks = 0; ks < HDIM; ks += 32) {
    const v16bf a = load_a_frag_f32(Xa, m0, HDIM, ks, lane, na - 1);
#pragma unroll
    for (int t = 0; t < 8; ++t) {
      const v16bf b = load_b_frag_bf(Wpb, t * 16, HDIM, ks, lane);
      acc[t] = __builtin_amdgcn_wmma_f32_16x16x32_bf16(
          false, a, false, b, (short)0, acc[t], false, false);
    }
  }

  // ---- spill proj tile to LDS as bf16 (row-major 16 x 128) ----
  __bf16* L = lds + (size_t)wave * 16 * HDIM;
#pragma unroll
  for (int t = 0; t < 8; ++t) {
    const int n = t * 16 + (lane & 15);
#pragma unroll
    for (int i = 0; i < 8; ++i) {
      const int m = i + 8 * half;
      L[m * HDIM + n] = (__bf16)acc[t][i];
    }
  }
  __syncthreads();

  // ---- h = relu(proj @ Wc1^T + neigh @ Wc2^T + bc) ----
  v8f acc2[8];
#pragma unroll
  for (int t = 0; t < 8; ++t) {
    const float bv = bc[t * 16 + (lane & 15)];
#pragma unroll
    for (int i = 0; i < 8; ++i) acc2[t][i] = bv;
  }
  // first 128 of K: proj tile from LDS (two 16B ds loads per fragment)
#pragma unroll
  for (int ks = 0; ks < HDIM; ks += 32) {
    const v8bf* pa = (const v8bf*)(L + (size_t)(lane & 15) * HDIM + ks + half * 8);
    const v8bf alo = pa[0];
    const v8bf ahi = pa[2];           // +16 bf16 elements
    const v16bf a = __builtin_shufflevector(alo, ahi, 0, 1, 2, 3, 4, 5, 6, 7,
                                            8, 9, 10, 11, 12, 13, 14, 15);
#pragma unroll
    for (int t = 0; t < 8; ++t) {
      const v16bf b = load_b_frag_bf(Wcb, t * 16, 2 * HDIM, ks, lane);
      acc2[t] = __builtin_amdgcn_wmma_f32_16x16x32_bf16(
          false, a, false, b, (short)0, acc2[t], false, false);
    }
  }
  // second 128 of K: neigh_a from global (f32, converted once per element)
#pragma unroll
  for (int ks = 0; ks < HDIM; ks += 32) {
    const v16bf a = load_a_frag_f32(neigh, m0, HDIM, ks, lane, na - 1);
#pragma unroll
    for (int t = 0; t < 8; ++t) {
      const v16bf b = load_b_frag_bf(Wcb, t * 16, 2 * HDIM, HDIM + ks, lane);
      acc2[t] = __builtin_amdgcn_wmma_f32_16x16x32_bf16(
          false, a, false, b, (short)0, acc2[t], false, false);
    }
  }

  // ---- out = sigmoid(relu(h) . Wo + bo) ----
  float partial[8];
#pragma unroll
  for (int i = 0; i < 8; ++i) partial[i] = 0.f;
#pragma unroll
  for (int t = 0; t < 8; ++t) {
    const float wo = Wo[t * 16 + (lane & 15)];
#pragma unroll
    for (int i = 0; i < 8; ++i) {
      const float h = acc2[t][i] > 0.f ? acc2[t][i] : 0.f;
      partial[i] = fmaf(h, wo, partial[i]);
    }
  }
  const float b0 = bo[0];
#pragma unroll
  for (int i = 0; i < 8; ++i) {
    float v = partial[i];
    v += __shfl_xor(v, 1, 32);
    v += __shfl_xor(v, 2, 32);
    v += __shfl_xor(v, 4, 32);
    v += __shfl_xor(v, 8, 32);   // lanes 0-15 -> rowsum(i); lanes 16-31 -> rowsum(i+8)
    const int m = m0 + i + 8 * half;
    if ((lane & 15) == 0 && m < na) {
      out[m] = 1.f / (1.f + __expf(-(v + b0)));
    }
  }
}

// ---------------- Host launcher ---------------------------------------------

extern "C" void kernel_launch(void* const* d_in, const int* in_sizes, int n_in,
                              void* d_out, int out_size, void* d_ws, size_t ws_size,
                              hipStream_t stream) {
  const float* x_account  = (const float*)d_in[0];
  const float* x_merchant = (const float*)d_in[1];
  const int*   e_ma_src   = (const int*)d_in[5];
  const int*   e_ma_dst   = (const int*)d_in[6];
  const float* e_ma_w     = (const float*)d_in[7];
  const float* Wp_a = (const float*)d_in[8];
  const float* bp_a = (const float*)d_in[9];
  const float* Wr_ma = (const float*)d_in[14];
  const float* br_ma = (const float*)d_in[15];
  const float* Wc_a = (const float*)d_in[16];
  const float* bc_a = (const float*)d_in[17];
  const float* Wo   = (const float*)d_in[20];
  const float* bo   = (const float*)d_in[21];

  const int NA = in_sizes[0] / 128;   // 200000
  const int NM = in_sizes[1] / 64;    // 100000
  const int NE = in_sizes[5];         // 1000000

  // workspace layout
  float*  neigh = (float*)d_ws;                         // NA*128 f32 = 102.4 MB
  float*  lin   = (float*)d_ws + (size_t)NA * HDIM;     // NM*128 f32 = 51.2 MB
  __bf16* wbase = (__bf16*)(lin + (size_t)NM * HDIM);   // bf16 weights (~114 KB)
  __bf16* Wp_bf = wbase;                                // 128*128
  __bf16* Wr_bf = Wp_bf + 128 * 128;                    // 128*64
  __bf16* Wc_bf = Wr_bf + 128 * 64;                     // 128*256
  float*  outp  = (float*)d_out;

  // P) one-shot bf16 weight conversion (L2-resident afterwards)
  cvt_f32_bf16<<<dim3((128 * 128 + 255) / 256), dim3(256), 0, stream>>>(Wp_a, Wp_bf, 128 * 128);
  cvt_f32_bf16<<<dim3((128 * 64  + 255) / 256), dim3(256), 0, stream>>>(Wr_ma, Wr_bf, 128 * 64);
  cvt_f32_bf16<<<dim3((128 * 256 + 255) / 256), dim3(256), 0, stream>>>(Wc_a, Wc_bf, 128 * 256);

  // 0) zero the scatter accumulator
  const int n4 = (NA * HDIM) / 4;
  zero_f4<<<dim3((n4 + 255) / 256), dim3(256), 0, stream>>>((float4*)neigh, n4);

  // 1) lin_ma GEMM (bias fused)
  const int t1 = (NM + 15) / 16;
  lin_ma_kernel<<<dim3((t1 + 7) / 8), dim3(256), 0, stream>>>(
      x_merchant, Wr_bf, br_ma, lin, NM);

  // 2) edge scatter-add into neigh_a
  scatter_kernel<<<dim3((NE + 7) / 8), dim3(256), 0, stream>>>(
      e_ma_src, e_ma_dst, e_ma_w, lin, neigh, NE);

  // 3) fused proj + combine + head
  const int t2 = (NA + 15) / 16;
  account_fused<<<dim3((t2 + 7) / 8), dim3(256), 0, stream>>>(
      x_account, Wp_bf, bp_a, Wc_bf, bc_a, Wo, bo, neigh, outp, NA);
}
